// HSMM_4243427689001
// MI455X (gfx1250) — matrix operand order repaired and checked by reading, not compiled, not run on hardware
//
#include <hip/hip_runtime.h>
#include <hip/hip_bf16.h>
#include <math.h>

typedef __attribute__((ext_vector_type(16))) _Float16 v16h;
typedef __attribute__((ext_vector_type(8)))  float    v8f;

#define NEGV (-1e30f)
#define Bb 32
#define Tt 256
#define Ee 128
#define Qq 20
#define Kk 50
#define Ll 8
#define Ss 258            // T + 2
#define ROWS (Ss*Bb)      // 8256 rows for input GEMM
#define GJ 80             // 4*Q gate width per direction

// workspace byte offsets (all 256B aligned)
#define OFF_XSH   ((size_t)0)                    // 8256*128 half  = 2,113,536 B
#define OFF_WIHT  ((size_t)2113536)              // 160*128 half   = 40,960 B (both dirs, col-major)
#define OFF_W1T   (OFF_WIHT + 40960)             // 32*64 half (zero-padded, col-major)
#define OFF_W2T   (OFF_W1T + 4096)               // 64*32 half (zero-padded, col-major)
#define OFF_GX    (OFF_W2T + 4096)               // 2*8256*80 f32  = 5,283,840 B
#define OFF_HF    (OFF_GX + 5283840)             // 258*32*20 f32  = 660,480 B
#define OFF_HB    (OFF_HF + 660480)

union Frag { v16h v; uint4 u[2]; };

static __device__ inline v8f wmma16x16x32(v16h a, v16h b, v8f c) {
  return __builtin_amdgcn_wmma_f32_16x16x32_f16(false, a, false, b, (short)0, c, false, false);
}
static __device__ inline float sigm(float x) { return 1.0f / (1.0f + __expf(-x)); }

// ---------------- K1a: embedding gather + BOS/EOS pad, fp32 -> fp16 ----------------
__global__ void hsmm_embed(const int* __restrict__ x, const float* __restrict__ emb,
                           _Float16* __restrict__ xsH) {
  int idx = blockIdx.x * blockDim.x + threadIdx.x;    // over Ss*Bb*Ee
  if (idx >= Ss * Bb * Ee) return;
  int e = idx & (Ee - 1);
  int b = (idx >> 7) & (Bb - 1);
  int s = idx >> 12;
  int tok = (s == 0) ? 2 : (s == Ss - 1) ? 3 : x[b * Tt + (s - 1)];
  xsH[idx] = (_Float16)emb[tok * Ee + e];
}

// ---------------- K1b: pack weights col-major fp16 (zero padded) ----------------
__global__ void hsmm_packw(const float* __restrict__ Wihf, const float* __restrict__ Wihb,
                           const float* __restrict__ W1,   const float* __restrict__ W2,
                           _Float16* __restrict__ WihT, _Float16* __restrict__ W1T,
                           _Float16* __restrict__ W2T) {
  int idx = blockIdx.x * blockDim.x + threadIdx.x;
  if (idx < 160 * 128) {                 // WihT[j][e], j in [0,160): fwd 0..79, bwd 80..159
    int j = idx >> 7, e = idx & 127;
    float v = (j < GJ) ? Wihf[e * GJ + j] : Wihb[e * GJ + (j - GJ)];
    WihT[idx] = (_Float16)v;
  } else if (idx < 160 * 128 + 32 * 64) {  // W1T[c][k], c<20 valid, k<40 valid
    int i2 = idx - 160 * 128;
    int c = i2 >> 6, k = i2 & 63;
    W1T[i2] = (c < Qq && k < 2 * Qq) ? (_Float16)W1[k * Qq + c] : (_Float16)0.0f;
  } else if (idx < 160 * 128 + 32 * 64 + 64 * 32) { // W2T[c][k], c<50, k<20
    int i3 = idx - 160 * 128 - 32 * 64;
    int c = i3 >> 5, k = i3 & 31;
    W2T[i3] = (c < Kk && k < Qq) ? (_Float16)W2[k * Kk + c] : (_Float16)0.0f;
  }
}

// ---------------- K2: gates_x = xs @ [W_ih_f | W_ih_b] via WMMA f16 ----------------
// rows r = s*32+b (8256 = 516 M-tiles), K = 128 (4 steps), N = 160 (10 tiles)
__global__ void hsmm_gates_gemm(const _Float16* __restrict__ xsH,
                                const _Float16* __restrict__ WihT,
                                float* __restrict__ gatesX) {
  int lane = threadIdx.x & 31;
  int mtile = blockIdx.x * (blockDim.x >> 5) + (threadIdx.x >> 5);
  if (mtile >= ROWS / 16) return;
  int rowA = mtile * 16 + (lane & 15);
  int kb8 = (lane >> 4) * 8;

  Frag a[4];
#pragma unroll
  for (int ks = 0; ks < 4; ++ks) {
    const _Float16* p = xsH + rowA * 128 + ks * 32 + kb8;
    a[ks].u[0] = *(const uint4*)(p);
    a[ks].u[1] = *(const uint4*)(p + 16);
  }
  int kb16 = (lane >> 4) * 8;
#pragma unroll
  for (int n = 0; n < 10; ++n) {
    int col = n * 16 + (lane & 15);
    v8f acc = {};
#pragma unroll
    for (int ks = 0; ks < 4; ++ks) {
      Frag bfr;
      const _Float16* p = WihT + col * 128 + ks * 32 + kb16;
      bfr.u[0] = *(const uint4*)(p);
      bfr.u[1] = *(const uint4*)(p + 16);
      acc = wmma16x16x32(a[ks].v, bfr.v, acc);
    }
    int dir = col / GJ, j = col % GJ;
    float* g = gatesX + (size_t)dir * ROWS * GJ;
#pragma unroll
    for (int r = 0; r < 8; ++r) {
      int rowg = mtile * 16 + r + 8 * (lane >> 4);
      g[rowg * GJ + j] = acc[r];
    }
  }
}

// ---------------- K3: sequential LSTM scan (one block per direction) ----------------
__global__ __launch_bounds__(320) void hsmm_lstm(const float* __restrict__ gatesX,
                          const float* __restrict__ Whhf, const float* __restrict__ bf,
                          const float* __restrict__ Whhb, const float* __restrict__ bb,
                          float* __restrict__ hfS, float* __restrict__ hbS) {
  __shared__ float sW[Qq][GJ];
  __shared__ float sB[GJ];
  __shared__ float sH[Bb][Qq];
  __shared__ float sC[Bb][Qq];
  int d = blockIdx.x;
  const float* Whh = d ? Whhb : Whhf;
  const float* bia = d ? bb : bf;
  const float* gx  = gatesX + (size_t)d * ROWS * GJ;
  float* hs = d ? hbS : hfS;
  int tid = threadIdx.x;
  for (int i = tid; i < Qq * GJ; i += 320) sW[i / GJ][i % GJ] = Whh[i];
  if (tid < GJ) sB[tid] = bia[tid];
  for (int i = tid; i < Bb * Qq; i += 320) { sH[i / Qq][i % Qq] = 0.f; sC[i / Qq][i % Qq] = 0.f; }
  __syncthreads();

  for (int step = 0; step < Ss; ++step) {
    int s = d ? (Ss - 1 - step) : step;
    float hN[2], cN[2];
#pragma unroll
    for (int cpart = 0; cpart < 2; ++cpart) {
      int cell = tid + cpart * 320;
      int b = cell / Qq, q = cell % Qq;
      const float* gr = gx + (size_t)(s * Bb + b) * GJ;
      float ai = sB[q]          + gr[q];
      float af = sB[Qq + q]     + gr[Qq + q];
      float ag = sB[2 * Qq + q] + gr[2 * Qq + q];
      float ao = sB[3 * Qq + q] + gr[3 * Qq + q];
#pragma unroll
      for (int qp = 0; qp < Qq; ++qp) {
        float hv = sH[b][qp];
        ai = fmaf(hv, sW[qp][q], ai);
        af = fmaf(hv, sW[qp][Qq + q], af);
        ag = fmaf(hv, sW[qp][2 * Qq + q], ag);
        ao = fmaf(hv, sW[qp][3 * Qq + q], ao);
      }
      float c = sigm(af) * sC[b][q] + sigm(ai) * tanhf(ag);
      float h = sigm(ao) * tanhf(c);
      cN[cpart] = c; hN[cpart] = h;
    }
    __syncthreads();
#pragma unroll
    for (int cpart = 0; cpart < 2; ++cpart) {
      int cell = tid + cpart * 320;
      int b = cell / Qq, q = cell % Qq;
      sH[b][q] = hN[cpart]; sC[b][q] = cN[cpart];
      hs[(size_t)s * (Bb * Qq) + b * Qq + q] = hN[cpart];
    }
    __syncthreads();
  }
}

// ---------------- K4: span-score MLP via WMMA (only the L=8 needed bands) ----------------
// tile = ((t*8 + l)*2 + bt); 16 rows = 16 consecutive batch entries of span (start..t)
__global__ void hsmm_score(const float* __restrict__ hfS, const float* __restrict__ hbS,
                           const _Float16* __restrict__ W1T, const float* __restrict__ b1,
                           const _Float16* __restrict__ W2T, const float* __restrict__ b2,
                           float* __restrict__ out) {
  __shared__ __align__(16) _Float16 catS[8][16][64];
  __shared__ __align__(16) _Float16 hidS[8][16][32];
  int wave = threadIdx.x >> 5;
  int lane = threadIdx.x & 31;
  int tile = blockIdx.x * 8 + wave;
  int t   = tile >> 4;
  int lab = (tile >> 1) & 7;
  int bt  = tile & 1;
  int start = t - (Ll - 1) + lab;
  float* outBase = out + ((size_t)(lab * Tt + t) * Bb + bt * 16) * Kk;

  if (start < 0) {                 // invalid span: fill NEG, no compute
    for (int idx = lane; idx < 16 * Kk; idx += 32)
      outBase[(idx / Kk) * Kk + (idx % Kk)] = NEGV;
    return;
  }

  // build cat[16][64] = [ hf[t+2]-hf[start+1] | hb[start]-hb[t+1] | 0 pad ] per batch row
  for (int idx = lane; idx < 16 * 64; idx += 32) {
    int r = idx >> 6, k = idx & 63;
    int b = bt * 16 + r;
    float v = 0.f;
    if (k < Qq)
      v = hfS[(size_t)(t + 2) * 640 + b * Qq + k] - hfS[(size_t)(start + 1) * 640 + b * Qq + k];
    else if (k < 2 * Qq)
      v = hbS[(size_t)start * 640 + b * Qq + (k - Qq)] - hbS[(size_t)(t + 1) * 640 + b * Qq + (k - Qq)];
    catS[wave][r][k] = (_Float16)v;
  }
  asm volatile("s_wait_dscnt 0" ::: "memory");   // wave-private staging: order DS, no block barrier

  int rA = lane & 15;
  int kb8 = (lane >> 4) * 8;
  int kb16 = (lane >> 4) * 16;

  // layer 1: hidden = tanh(cat @ W1 + b1), N padded 20->32, K padded 40->64
#pragma unroll
  for (int n = 0; n < 2; ++n) {
    int col = n * 16 + (lane & 15);
    v8f acc = {};
#pragma unroll
    for (int ks = 0; ks < 2; ++ks) {
      Frag a, bfr;
      const _Float16* pa = &catS[wave][rA][ks * 32 + kb8];
      a.u[0] = *(const uint4*)(pa);
      a.u[1] = *(const uint4*)(pa + 16);
      const _Float16* pb = W1T + col * 64 + ks * 32 + kb8;
      bfr.u[0] = *(const uint4*)(pb);
      bfr.u[1] = *(const uint4*)(pb + 16);
      acc = wmma16x16x32(a.v, bfr.v, acc);
    }
    float bias = (col < Qq) ? b1[col] : 0.f;
#pragma unroll
    for (int r = 0; r < 8; ++r)
      hidS[wave][r + 8 * (lane >> 4)][col] = (_Float16)tanhf(acc[r] + bias);
  }
  asm volatile("s_wait_dscnt 0" ::: "memory");

  // layer 2: scores = hidden @ W2 + b2, K padded 20->32, N padded 50->64
  Frag a2;
  {
    const _Float16* pa = &hidS[wave][rA][kb8];
    a2.u[0] = *(const uint4*)(pa);
    a2.u[1] = *(const uint4*)(pa + 16);
  }
#pragma unroll
  for (int n2 = 0; n2 < 4; ++n2) {
    int col = n2 * 16 + (lane & 15);
    Frag bfr;
    const _Float16* pb = W2T + col * 32 + kb8;
    bfr.u[0] = *(const uint4*)(pb);
    bfr.u[1] = *(const uint4*)(pb + 16);
    v8f c0 = {};
    v8f acc = wmma16x16x32(a2.v, bfr.v, c0);
    if (col < Kk) {
      float bias = b2[col];
#pragma unroll
      for (int r = 0; r < 8; ++r)
        outBase[(size_t)(r + 8 * (lane >> 4)) * Kk + col] = acc[r] + bias;
    }
  }
}

extern "C" void kernel_launch(void* const* d_in, const int* in_sizes, int n_in,
                              void* d_out, int out_size, void* d_ws, size_t ws_size,
                              hipStream_t stream) {
  const int*   x     = (const int*)  d_in[0];
  const float* emb   = (const float*)d_in[1];
  const float* Wihf  = (const float*)d_in[2];
  const float* Whhf  = (const float*)d_in[3];
  const float* bf    = (const float*)d_in[4];
  const float* Wihb  = (const float*)d_in[5];
  const float* Whhb  = (const float*)d_in[6];
  const float* bbv   = (const float*)d_in[7];
  const float* W1    = (const float*)d_in[8];
  const float* b1    = (const float*)d_in[9];
  const float* W2    = (const float*)d_in[10];
  const float* b2    = (const float*)d_in[11];
  float* out = (float*)d_out;

  char* ws = (char*)d_ws;
  _Float16* xsH  = (_Float16*)(ws + OFF_XSH);
  _Float16* WihT = (_Float16*)(ws + OFF_WIHT);
  _Float16* W1T  = (_Float16*)(ws + OFF_W1T);
  _Float16* W2T  = (_Float16*)(ws + OFF_W2T);
  float*    gX   = (float*)   (ws + OFF_GX);
  float*    hfS  = (float*)   (ws + OFF_HF);
  float*    hbS  = (float*)   (ws + OFF_HB);

  // K1: embedding gather + weight packing
  hsmm_embed <<<(Ss * Bb * Ee + 255) / 256, 256, 0, stream>>>(x, emb, xsH);
  hsmm_packw <<<(160 * 128 + 32 * 64 + 64 * 32 + 255) / 256, 256, 0, stream>>>(
      Wihf, Wihb, W1, W2, WihT, W1T, W2T);

  // K2: input-projection GEMM, both directions fused (N=160)
  hsmm_gates_gemm<<<(ROWS / 16 + 7) / 8, 256, 0, stream>>>(xsH, WihT, gX);

  // K3: sequential bidirectional LSTM scan
  hsmm_lstm<<<2, 320, 0, stream>>>(gX, Whhf, bf, Whhb, bbv, hfS, hbS);

  // K4: banded span scorer (8 WMMAs per 16-row tile)
  hsmm_score<<<(Tt * Ll * 2) / 8, 256, 0, stream>>>(hfS, hbS, W1T, b1, W2T, b2, out);
}
